// MultiheadAttention_87522843558520
// MI455X (gfx1250) — matrix operand order repaired and checked by reading, not compiled, not run on hardware
//
#include <hip/hip_runtime.h>
#include <hip/hip_bf16.h>

typedef __attribute__((ext_vector_type(16))) _Float16 v16h;
typedef __attribute__((ext_vector_type(8)))  _Float16 v8h;
typedef __attribute__((ext_vector_type(8)))  float    v8f;
typedef __attribute__((ext_vector_type(4)))  float    v4f;

#define T_LEN 2048
#define BSZ   2
#define EMB   1024
#define NH    16
#define HD    64
#define ROWS  (T_LEN*BSZ)      // 4096
#define WELEMS (EMB*EMB)       // 1048576

static __device__ __forceinline__ v8f wmma_f16(v16h a, v16h b, v8f c) {
  // D = A(16x32 f16) x B(32x16 f16) + C(16x16 f32)
  return __builtin_amdgcn_wmma_f32_16x16x32_f16(false, a, false, b, (short)0, c, false, false);
}

// B operand: lane holds column N=lane%16, K chunks {0..7} and {16..23} (two 16B chunks at +0,+16; caller folds +8 for hi half)
static __device__ __forceinline__ v16h load_b16(const _Float16* p) {
  v16h r;
  ((v8h*)&r)[0] = *(const v8h*)(p);
  ((v8h*)&r)[1] = *(const v8h*)(p + 16);
  return r;
}
// A operand (f16 source): lane holds row M=lane%16, K chunks {0..7} and {16..23} (caller folds +8 for hi half)
static __device__ __forceinline__ v16h load_a16(const _Float16* p) {
  v16h r;
  ((v8h*)&r)[0] = *(const v8h*)(p);
  ((v8h*)&r)[1] = *(const v8h*)(p + 16);
  return r;
}
// A operand from f32 source, converting to f16
static __device__ __forceinline__ v16h load_a32(const float* p) {
  v4f x0 = *(const v4f*)(p);
  v4f x1 = *(const v4f*)(p + 4);
  v4f x2 = *(const v4f*)(p + 16);
  v4f x3 = *(const v4f*)(p + 20);
  v16h r;
#pragma unroll
  for (int i = 0; i < 4; ++i) {
    r[i]      = (_Float16)x0[i];
    r[4 + i]  = (_Float16)x1[i];
    r[8 + i]  = (_Float16)x2[i];
    r[12 + i] = (_Float16)x3[i];
  }
  return r;
}

// ---------------------------------------------------------------------------
// Prep: convert the four weight matrices to f16, mask -> additive score bias.
// ---------------------------------------------------------------------------
__global__ void mha_prep_kernel(const float* __restrict__ wq, const float* __restrict__ wk,
                                const float* __restrict__ wv, const float* __restrict__ wo,
                                const unsigned char* __restrict__ mask,
                                _Float16* __restrict__ w16, float* __restrict__ sbias) {
  int i = blockIdx.x * 256 + threadIdx.x;
  if (i < WELEMS) {
    w16[i]              = (_Float16)wq[i];
    w16[WELEMS + i]     = (_Float16)wk[i];
    w16[2 * WELEMS + i] = (_Float16)wv[i];
    w16[3 * WELEMS + i] = (_Float16)wo[i];
  }
  if (i < BSZ * T_LEN) sbias[i] = mask[i] ? -__builtin_inff() : 0.0f;
}

// ---------------------------------------------------------------------------
// Fused QKV projection: out = X @ W^T + b  (q additionally scaled by D^-0.5)
// One wave computes a 32(M) x 64(N) tile. grid.z selects q/k/v.
// Outputs: q16,k16 in [b][h][t][d]; v transposed to [b][h][d][t].
// ---------------------------------------------------------------------------
__global__ void mha_qkv_kernel(const float* __restrict__ xq, const float* __restrict__ xk,
                               const float* __restrict__ xv, const _Float16* __restrict__ w16,
                               const float* __restrict__ bq, const float* __restrict__ bk,
                               const float* __restrict__ bv,
                               _Float16* __restrict__ q16, _Float16* __restrict__ k16,
                               _Float16* __restrict__ vt16) {
  const int lane = threadIdx.x;
  const int lm = lane & 15, hi = lane >> 4;
  const int mz = blockIdx.x, nz = blockIdx.y, z = blockIdx.z;
  const float* X = (z == 0) ? xq : (z == 1) ? xk : xv;
  const _Float16* W = w16 + (size_t)z * WELEMS;
  const float* bias = (z == 0) ? bq : (z == 1) ? bk : bv;
  const int offA = hi * 8, offB = hi * 8;

  v8f acc[2][4] = {};
  for (int k = 0; k < EMB; k += 32) {
    v16h a0 = load_a32(X + (size_t)(32 * mz + lm) * EMB + k + offA);
    v16h a1 = load_a32(X + (size_t)(32 * mz + 16 + lm) * EMB + k + offA);
#pragma unroll
    for (int j = 0; j < 4; ++j) {
      v16h bm = load_b16(W + (size_t)(64 * nz + 16 * j + lm) * EMB + k + offB);
      acc[0][j] = wmma_f16(a0, bm, acc[0][j]);
      acc[1][j] = wmma_f16(a1, bm, acc[1][j]);
    }
  }

  const float scl = (z == 0) ? 0.125f : 1.0f;  // D^-0.5 = 1/8
#pragma unroll
  for (int j = 0; j < 4; ++j) {
    const int n = 64 * nz + 16 * j + lm;
    const float bvv = bias[n];
    const int h = n >> 6, d = n & 63;
#pragma unroll
    for (int mi = 0; mi < 2; ++mi) {
#pragma unroll
      for (int jj = 0; jj < 8; ++jj) {
        const int m = 32 * mz + 16 * mi + jj + 8 * hi;   // row index = t*B + b
        const float val = (acc[mi][j][jj] + bvv) * scl;
        const int t = m >> 1, bb = m & 1;                 // B == 2
        if (z == 0)
          q16[((size_t)(bb * NH + h) * T_LEN + t) * HD + d] = (_Float16)val;
        else if (z == 1)
          k16[((size_t)(bb * NH + h) * T_LEN + t) * HD + d] = (_Float16)val;
        else
          vt16[((size_t)(bb * NH + h) * HD + d) * T_LEN + t] = (_Float16)val;
      }
    }
  }
}

// ---------------------------------------------------------------------------
// Flash attention. One wave per (bh, 16-query tile). Computes S^T = K @ Q^T so
// the S^T accumulator layout (lane = query) is exactly the A-operand layout
// for P @ V after a per-lane f32->f16 pack (no cross-lane shuffles needed).
// ---------------------------------------------------------------------------
__global__ void mha_attn_kernel(const _Float16* __restrict__ q16, const _Float16* __restrict__ k16,
                                const _Float16* __restrict__ vt16, const float* __restrict__ sbias,
                                _Float16* __restrict__ attn16) {
  const int lane = threadIdx.x;
  const int lm = lane & 15, hi = lane >> 4;
  const int qt = blockIdx.x;          // 16-query tile
  const int bh = blockIdx.y;
  const int b = bh >> 4, hh = bh & 15;
  const _Float16* Qp = q16 + (size_t)bh * T_LEN * HD;
  const _Float16* Kp = k16 + (size_t)bh * T_LEN * HD;
  const _Float16* Vp = vt16 + (size_t)bh * HD * T_LEN;
  const float* sb = sbias + b * T_LEN;
  const int offA = hi * 8, offB = hi * 8;

  // Q as B-operand (lane = query column), loaded once: d in [0,32) and [32,64)
  const int qrow = 16 * qt + lm;
  v16h qb[2];
#pragma unroll
  for (int s = 0; s < 2; ++s) qb[s] = load_b16(Qp + (size_t)qrow * HD + 32 * s + offB);

  v8f o[4] = {};                       // O: 16 queries x 64 d
  float mrun = -__builtin_inff(), lrun = 0.0f;

  for (int kb = 0; kb < T_LEN; kb += 128) {
    // ---- S^T tiles: 8 x (16 keys x 16 queries) ----
    v8f st[8];
#pragma unroll
    for (int jt = 0; jt < 8; ++jt) {
      const int kt = kb + 16 * jt;
      v8f sa = {};
#pragma unroll
      for (int s = 0; s < 2; ++s) {
        v16h ka = load_a16(Kp + (size_t)(kt + lm) * HD + 32 * s + offA);
        sa = wmma_f16(ka, qb[s], sa);
      }
      // additive key-padding bias: slot jj <-> key kt + jj + 8*hi
      const float* bp = sb + kt + 8 * hi;
      v4f b0 = *(const v4f*)(bp);
      v4f b1 = *(const v4f*)(bp + 4);
#pragma unroll
      for (int jj = 0; jj < 4; ++jj) { sa[jj] += b0[jj]; sa[4 + jj] += b1[jj]; }
      st[jt] = sa;
    }

    // ---- online softmax: all of a lane's 64 values share one query row ----
    float bm = -__builtin_inff();
#pragma unroll
    for (int jt = 0; jt < 8; ++jt)
#pragma unroll
      for (int jj = 0; jj < 8; ++jj) bm = fmaxf(bm, st[jt][jj]);
    bm = fmaxf(bm, __shfl_xor(bm, 16, 32));
    const float mnew = fmaxf(mrun, bm);
    const float corr = (mrun == -__builtin_inff()) ? 0.0f : __expf(mrun - mnew);
    mrun = mnew;
    lrun *= corr;

    // rescale O: O-slot jj corresponds to query row jj + 8*hi -> fetch that row's corr
#pragma unroll
    for (int jj = 0; jj < 8; ++jj) {
      const float c = __shfl(corr, hi ? (jj + 8) : jj, 32);
#pragma unroll
      for (int dt = 0; dt < 4; ++dt) o[dt][jj] *= c;
    }

    float rs = 0.0f;
#pragma unroll
    for (int jt = 0; jt < 8; ++jt)
#pragma unroll
      for (int jj = 0; jj < 8; ++jj) {
        const float e = __expf(st[jt][jj] - mnew);
        st[jt][jj] = e;
        rs += e;
      }
    rs += __shfl_xor(rs, 16, 32);
    lrun += rs;

    // ---- O += P @ V : P A-operand comes straight from st tile pairs ----
#pragma unroll
    for (int ks = 0; ks < 4; ++ks) {
      v16h pa;
#pragma unroll
      for (int i = 0; i < 8; ++i) {
        pa[i]     = (_Float16)st[2 * ks][i];
        pa[8 + i] = (_Float16)st[2 * ks + 1][i];
      }
      const int keyc = kb + 32 * ks + offB;
#pragma unroll
      for (int dt = 0; dt < 4; ++dt) {
        v16h vb = load_b16(Vp + (size_t)(16 * dt + lm) * T_LEN + keyc);
        o[dt] = wmma_f16(pa, vb, o[dt]);
      }
    }
  }

  // ---- normalize and write attn rows in (t*B+b, h*64+d) f16 layout ----
#pragma unroll
  for (int jj = 0; jj < 8; ++jj) {
    const float lv = __shfl(lrun, hi ? (jj + 8) : jj, 32);
    const float inv = 1.0f / lv;
    const int tq = 16 * qt + jj + 8 * hi;
#pragma unroll
    for (int dt = 0; dt < 4; ++dt) {
      attn16[(size_t)(tq * BSZ + b) * EMB + hh * HD + 16 * dt + lm] =
          (_Float16)(o[dt][jj] * inv);
    }
  }
}

// ---------------------------------------------------------------------------
// Output projection: out = attn @ wo^T + bo (f32 result).
// ---------------------------------------------------------------------------
__global__ void mha_oproj_kernel(const _Float16* __restrict__ attn16,
                                 const _Float16* __restrict__ wo16,
                                 const float* __restrict__ bo, float* __restrict__ out) {
  const int lane = threadIdx.x;
  const int lm = lane & 15, hi = lane >> 4;
  const int mz = blockIdx.x, nz = blockIdx.y;
  const int offA = hi * 8, offB = hi * 8;

  v8f acc[2][4] = {};
  for (int k = 0; k < EMB; k += 32) {
    v16h a0 = load_a16(attn16 + (size_t)(32 * mz + lm) * EMB + k + offA);
    v16h a1 = load_a16(attn16 + (size_t)(32 * mz + 16 + lm) * EMB + k + offA);
#pragma unroll
    for (int j = 0; j < 4; ++j) {
      v16h bm = load_b16(wo16 + (size_t)(64 * nz + 16 * j + lm) * EMB + k + offB);
      acc[0][j] = wmma_f16(a0, bm, acc[0][j]);
      acc[1][j] = wmma_f16(a1, bm, acc[1][j]);
    }
  }
#pragma unroll
  for (int j = 0; j < 4; ++j) {
    const int n = 64 * nz + 16 * j + lm;
    const float bb = bo[n];
#pragma unroll
    for (int mi = 0; mi < 2; ++mi) {
#pragma unroll
      for (int jj = 0; jj < 8; ++jj) {
        const int m = 32 * mz + 16 * mi + jj + 8 * hi;
        out[(size_t)m * EMB + n] = acc[mi][j][jj] + bb;
      }
    }
  }
}

extern "C" void kernel_launch(void* const* d_in, const int* in_sizes, int n_in,
                              void* d_out, int out_size, void* d_ws, size_t ws_size,
                              hipStream_t stream) {
  const float* query = (const float*)d_in[0];
  const float* key   = (const float*)d_in[1];
  const float* value = (const float*)d_in[2];
  const unsigned char* mask = (const unsigned char*)d_in[3];
  const float* wq = (const float*)d_in[4];
  const float* bq = (const float*)d_in[5];
  const float* wk = (const float*)d_in[6];
  const float* bk = (const float*)d_in[7];
  const float* wv = (const float*)d_in[8];
  const float* bv = (const float*)d_in[9];
  const float* wo = (const float*)d_in[10];
  const float* bo = (const float*)d_in[11];
  float* out = (float*)d_out;

  // Workspace layout (f16 units): 4 weight mats | q | k | v^T | attn | sbias(f32)
  _Float16* w16    = (_Float16*)d_ws;
  _Float16* q16    = w16 + (size_t)4 * WELEMS;
  _Float16* k16    = q16 + (size_t)ROWS * EMB;
  _Float16* vt16   = k16 + (size_t)ROWS * EMB;
  _Float16* attn16 = vt16 + (size_t)ROWS * EMB;
  float* sbias     = (float*)(attn16 + (size_t)ROWS * EMB);

  mha_prep_kernel<<<WELEMS / 256, 256, 0, stream>>>(wq, wk, wv, wo, mask, w16, sbias);
  mha_qkv_kernel<<<dim3(ROWS / 32, EMB / 64, 3), 32, 0, stream>>>(
      query, key, value, w16, bq, bk, bv, q16, k16, vt16);
  mha_attn_kernel<<<dim3(T_LEN / 16, BSZ * NH), 32, 0, stream>>>(
      q16, k16, vt16, sbias, attn16);
  mha_oproj_kernel<<<dim3(ROWS / 32, EMB / 64), 32, 0, stream>>>(
      attn16, w16 + (size_t)3 * WELEMS, bo, out);
}
